// MultiHeadAttention_35742717838015
// MI455X (gfx1250) — matrix operand order repaired and checked by reading, not compiled, not run on hardware
//
#include <hip/hip_runtime.h>
#include <stdint.h>

#define NH    16
#define HD    64
#define EMB   1024
#define BATCH 2
#define TSEQ  2048
#define KTOP  204   // max(1, int(0.1 * 2048))

typedef __bf16 bf16;
typedef __attribute__((ext_vector_type(16))) __bf16 v16bf;
typedef __attribute__((ext_vector_type(8)))  __bf16 v8bf;
typedef __attribute__((ext_vector_type(8)))  float  v8f;

__device__ __forceinline__ v16bf cat8(v8bf lo, v8bf hi) {
  return __builtin_shufflevector(lo, hi, 0,1,2,3,4,5,6,7,8,9,10,11,12,13,14,15);
}

// ---------------------------------------------------------------- fp32 -> bf16
__global__ void pack_bf16_kernel(const float* __restrict__ src,
                                 bf16* __restrict__ dst, int n) {
  int i = blockIdx.x * blockDim.x + threadIdx.x;
  if (i < n) dst[i] = (bf16)src[i];
}

// --------------------------------- fp32 [H,R,C] -> bf16 [H,C,R] (transpose)
__global__ void pack_transpose_kernel(const float* __restrict__ src,
                                      bf16* __restrict__ dst,
                                      int H, int R, int C) {
  size_t i = (size_t)blockIdx.x * blockDim.x + threadIdx.x;
  size_t total = (size_t)H * R * C;
  if (i >= total) return;
  int c = (int)(i % C);
  size_t t = i / C;
  int r = (int)(t % R);
  int h = (int)(t / R);
  dst[((size_t)h * C + c) * R + r] = (bf16)src[i];
}

// ------------------------------------------------- QKV projection (WMMA GEMM)
// One wave computes a 16(t) x 64(d) tile. WT is pre-transposed [H, D, E] so
// every B fragment is a contiguous 32B load. V is written transposed [BH,D,T].
// grid: x = T/16, z = bh*3 + which ; block = 32 (one wave)
__global__ void __launch_bounds__(32)
qkv_wmma_kernel(const bf16* __restrict__ Xq, const bf16* __restrict__ Xm,
                const bf16* __restrict__ WqT, const bf16* __restrict__ WkT,
                const bf16* __restrict__ WvT,
                bf16* __restrict__ Q, bf16* __restrict__ K, bf16* __restrict__ VT) {
  const int lane = threadIdx.x & 31;
  const int half = lane >> 4, lm = lane & 15;
  const int t0 = blockIdx.x * 16;
  const int z = blockIdx.z;
  const int which = z % 3, bh = z / 3;
  const int b = bh / NH, h = bh % NH;

  const bf16* X  = (which == 0) ? Xq : Xm;
  const bf16* WT = (which == 0) ? WqT : (which == 1) ? WkT : WvT;

  const bf16* xrow = X + ((size_t)b * TSEQ + t0 + lm) * EMB;
  const bf16* wbase = WT + ((size_t)h * HD + lm) * EMB;

  v8f acc[4] = {};
  for (int e0 = 0; e0 < EMB; e0 += 32) {
    __builtin_prefetch(xrow + e0 + 128, 0, 1);
    v16bf a = cat8(*(const v8bf*)(xrow + e0 + half * 16),
                   *(const v8bf*)(xrow + e0 + half * 16 + 8));
#pragma unroll
    for (int dt = 0; dt < 4; ++dt) {
      v16bf bm = *(const v16bf*)(wbase + (size_t)dt * 16 * EMB + e0 + half * 16);
      acc[dt] = __builtin_amdgcn_wmma_f32_16x16x32_bf16(false, a, false, bm,
                                                        (short)0, acc[dt], false, false);
    }
  }
  if (which < 2) {                       // Q, K row-major [BH, T, D]
    bf16* Y  = (which == 0) ? Q : K;
    bf16* yt = Y + ((size_t)bh * TSEQ + t0) * HD + lm;
#pragma unroll
    for (int dt = 0; dt < 4; ++dt)
#pragma unroll
      for (int r = 0; r < 8; ++r)
        yt[(size_t)(r + 8 * half) * HD + dt * 16] = (bf16)acc[dt][r];
  } else {                               // V transposed [BH, D, T]
    bf16* vt = VT + ((size_t)bh * HD + lm) * TSEQ + t0;
#pragma unroll
    for (int dt = 0; dt < 4; ++dt)
#pragma unroll
      for (int r = 0; r < 8; ++r)
        vt[(size_t)dt * 16 * TSEQ + r + 8 * half] = (bf16)acc[dt][r];
  }
}

// ------------------------------------------- scores = Q K^T / 8, causal mask
// One wave computes 16 queries x 64 keys (A fragment reused across 4 WMMAs).
// grid: x = T/16 (query tile), y = T/64 (key quad); block = 32
__global__ void __launch_bounds__(32)
scores_wmma_kernel(const bf16* __restrict__ Q, const bf16* __restrict__ K,
                   float* __restrict__ S, int bh) {
  const int q0 = blockIdx.x * 16, s0 = blockIdx.y * 64;
  if (s0 > q0 + 15) return;  // fully masked quad (uniform exit)
  const int lane = threadIdx.x & 31, half = lane >> 4, lm = lane & 15;
  const bf16* Qb = Q + (size_t)bh * TSEQ * HD;
  const bf16* Kb = K + (size_t)bh * TSEQ * HD;

  v8f acc[4] = {};
#pragma unroll
  for (int d0 = 0; d0 < HD; d0 += 32) {
    const bf16* qrow = Qb + (size_t)(q0 + lm) * HD + d0;
    v16bf a = cat8(*(const v8bf*)(qrow + half * 16),
                   *(const v8bf*)(qrow + half * 16 + 8));
#pragma unroll
    for (int kt = 0; kt < 4; ++kt) {
      v16bf bm = *(const v16bf*)(Kb + (size_t)(s0 + kt * 16 + lm) * HD + d0 + half * 16);
      acc[kt] = __builtin_amdgcn_wmma_f32_16x16x32_bf16(false, a, false, bm,
                                                        (short)0, acc[kt], false, false);
    }
  }
  const float scale = 0.125f;  // 64^-0.5
#pragma unroll
  for (int kt = 0; kt < 4; ++kt)
#pragma unroll
    for (int r = 0; r < 8; ++r) {
      int t = q0 + r + 8 * half, s = s0 + kt * 16 + lm;
      S[(size_t)t * TSEQ + s] = (s <= t) ? acc[kt][r] * scale : -__builtin_inff();
    }
}

// -------------- exact kth-largest per row: async row->LDS + 4-pass radix select
// Row (<=8KB) staged once in LDS via global_load_async_to_lds_b128 (ASYNCcnt),
// then 4 MSB-first 8-bit radix passes over order-preserving integer keys.
// grid: T blocks; block = 128
__global__ void __launch_bounds__(128)
topk_thresh_kernel(const float* __restrict__ S, float* __restrict__ thresh) {
  const int t = blockIdx.x;
  const int n = t + 1;
  const int keff = (KTOP < n) ? KTOP : n;
  const float* row = S + (size_t)t * TSEQ;
  const int tid = threadIdx.x;

  __shared__ __align__(16) float row_lds[TSEQ];
  __shared__ int hist[256];
  __shared__ unsigned sh_sel;
  __shared__ int sh_k;

  // stage row into LDS: L2 -> LDS without VGPR bounce (16B per async op)
  {
    const int nchunk = (n + 3) >> 2;  // 4 floats per b128 (tail over-read padded)
    for (int c = tid; c < nchunk; c += 128) {
      const float* gp = row + 4 * c;
      unsigned lds_off = (unsigned)(size_t)&row_lds[4 * c];  // low 32b = LDS offset
      asm volatile("global_load_async_to_lds_b128 %0, %1, off"
                   :: "v"(lds_off), "v"(gp) : "memory");
    }
    asm volatile("s_wait_asynccnt 0x0" ::: "memory");
  }
  __syncthreads();

  unsigned prefix = 0;
  int kk = keff;
  for (int shift = 24; shift >= 0; shift -= 8) {
    hist[tid] = 0;
    hist[tid + 128] = 0;
    __syncthreads();
    const unsigned himask = (shift == 24) ? 0u : (0xFFFFFFFFu << (shift + 8));
    for (int s = tid; s < n; s += 128) {
      unsigned u = __float_as_uint(row_lds[s]);
      unsigned key = (u & 0x80000000u) ? ~u : (u | 0x80000000u);
      if ((key & himask) == prefix)
        atomicAdd(&hist[(key >> shift) & 255], 1);
    }
    __syncthreads();
    if (tid == 0) {
      int acc = 0;
      int bkt = 255;
      for (; bkt > 0; --bkt) {
        int hcnt = hist[bkt];
        if (acc + hcnt >= kk) break;
        acc += hcnt;
      }
      sh_sel = (unsigned)bkt;
      sh_k = kk - acc;
    }
    __syncthreads();
    prefix |= sh_sel << shift;
    kk = sh_k;
    __syncthreads();
  }
  if (tid == 0) {
    unsigned u = (prefix & 0x80000000u) ? (prefix & 0x7FFFFFFFu) : ~prefix;
    thresh[t] = __uint_as_float(u);
  }
}

// ------------- filtered softmax stats + att @ V (WMMA), per 16-query tile
// VT is [BH, D, T] so every B fragment is a contiguous 32B load.
// grid: T/16 blocks; block = 256 (16 partials x 16 rows; wave 0 does WMMA)
__global__ void __launch_bounds__(256)
softmax_av_kernel(const float* __restrict__ S, const float* __restrict__ thresh,
                  const bf16* __restrict__ VT, bf16* __restrict__ O, int bh) {
  const int q0 = blockIdx.x * 16;
  const int tid = threadIdx.x;
  const int row = tid & 15, part = tid >> 4;
  const int b = bh / NH, h = bh % NH;

  __shared__ float redM[16][16];
  __shared__ float redS[16][16];
  __shared__ float rowM[16], rowS[16];

  // phase 1: streaming (max, sum of exp) with top-k threshold filter
  {
    const int t = q0 + row;
    const float th = thresh[t];
    const float* srow = S + (size_t)t * TSEQ;
    float m = -__builtin_inff(), sum = 0.f;
    for (int s = part; s <= t; s += 16) {
      float v = srow[s];
      if (v >= th) {
        if (v > m) { sum = sum * __expf(m - v) + 1.f; m = v; }
        else       { sum += __expf(v - m); }
      }
    }
    redM[part][row] = m;
    redS[part][row] = sum;
  }
  __syncthreads();
  for (int off = 8; off > 0; off >>= 1) {
    if (part < off) {
      float m1 = redM[part][row],       s1 = redS[part][row];
      float m2 = redM[part + off][row], s2 = redS[part + off][row];
      if (s2 > 0.f) {
        if (m2 > m1) { s1 = s1 * __expf(m1 - m2) + s2; m1 = m2; }
        else         { s1 += s2 * __expf(m2 - m1); }
      }
      redM[part][row] = m1; redS[part][row] = s1;
    }
    __syncthreads();
  }
  if (part == 0) { rowM[row] = redM[0][row]; rowS[row] = redS[0][row]; }
  __syncthreads();

  // phase 2: wave 0 rebuilds P as bf16 fragments and accumulates P @ V
  if (tid < 32) {
    const int lane = tid, half = lane >> 4, lm = lane & 15;
    const int tq = q0 + lm;                 // A-fragment row is lm for all elems
    const float m = rowM[lm];
    const float inv = (rowS[lm] > 0.f) ? 1.f / rowS[lm] : 0.f;
    const float thq = thresh[tq];
    const float* prow = S + (size_t)tq * TSEQ;
    const bf16* vb = VT + ((size_t)bh * HD + lm) * TSEQ;  // + dt*16*TSEQ

    v8f acc[4] = {};
    for (int s0 = 0; s0 <= q0 + 15; s0 += 32) {
      v16bf a;
#pragma unroll
      for (int j = 0; j < 8; ++j) {
        int s1 = s0 + half * 16 + j;
        int s2 = s0 + half * 16 + 8 + j;
        float v1 = prow[s1];
        float v2 = prow[s2];
        float p1 = (s1 <= tq && v1 >= thq) ? __expf(v1 - m) * inv : 0.f;
        float p2 = (s2 <= tq && v2 >= thq) ? __expf(v2 - m) * inv : 0.f;
        a[j] = (bf16)p1;
        a[j + 8] = (bf16)p2;
      }
#pragma unroll
      for (int dt = 0; dt < 4; ++dt) {
        v16bf bm = *(const v16bf*)(vb + (size_t)dt * 16 * TSEQ + s0 + half * 16);
        acc[dt] = __builtin_amdgcn_wmma_f32_16x16x32_bf16(false, a, false, bm,
                                                          (short)0, acc[dt], false, false);
      }
    }
    bf16* orow = O + ((size_t)b * TSEQ + q0) * (NH * HD) + h * HD + lm;
#pragma unroll
    for (int dt = 0; dt < 4; ++dt)
#pragma unroll
      for (int r = 0; r < 8; ++r)
        orow[(size_t)(r + 8 * half) * (NH * HD) + dt * 16] = (bf16)acc[dt][r];
  }
}

// ------------------------------------- output projection + bias (WMMA GEMM)
// One wave computes 16(row) x 64(e). WpT pre-transposed [E, F].
// grid: x = B*T/16, y = EMB/64; block = 32
__global__ void __launch_bounds__(32)
proj_wmma_kernel(const bf16* __restrict__ A, const bf16* __restrict__ WpT,
                 const float* __restrict__ bias, float* __restrict__ out) {
  const int lane = threadIdx.x & 31, half = lane >> 4, lm = lane & 15;
  const int r0 = blockIdx.x * 16, e0 = blockIdx.y * 64;
  const bf16* arow  = A + (size_t)(r0 + lm) * (NH * HD);
  const bf16* wbase = WpT + (size_t)(e0 + lm) * (NH * HD);

  v8f acc[4] = {};
  for (int f0 = 0; f0 < NH * HD; f0 += 32) {
    __builtin_prefetch(arow + f0 + 128, 0, 1);
    v16bf a = cat8(*(const v8bf*)(arow + f0 + half * 16),
                   *(const v8bf*)(arow + f0 + half * 16 + 8));
#pragma unroll
    for (int et = 0; et < 4; ++et) {
      v16bf bm = *(const v16bf*)(wbase + (size_t)et * 16 * (NH * HD) + f0 + half * 16);
      acc[et] = __builtin_amdgcn_wmma_f32_16x16x32_bf16(false, a, false, bm,
                                                        (short)0, acc[et], false, false);
    }
  }
#pragma unroll
  for (int et = 0; et < 4; ++et) {
    const float bv = bias[e0 + et * 16 + lm];
#pragma unroll
    for (int r = 0; r < 8; ++r)
      out[(size_t)(r0 + r + 8 * half) * EMB + e0 + et * 16 + lm] = acc[et][r] + bv;
  }
}

// ----------------------------------------------------------------- launcher
extern "C" void kernel_launch(void* const* d_in, const int* in_sizes, int n_in,
                              void* d_out, int out_size, void* d_ws, size_t ws_size,
                              hipStream_t stream) {
  const float* index  = (const float*)d_in[0];
  const float* memory = (const float*)d_in[1];
  const float* Wq     = (const float*)d_in[2];
  const float* Wk     = (const float*)d_in[3];
  const float* Wv     = (const float*)d_in[4];
  const float* Wp     = (const float*)d_in[5];
  const float* bp     = (const float*)d_in[6];
  float* out = (float*)d_out;

  char* ws = (char*)d_ws;
  size_t off = 0;
  auto alloc = [&](size_t bytes) -> void* {
    void* p = ws + off;
    off = (off + bytes + 255) & ~(size_t)255;
    return p;
  };

  const size_t nAct = (size_t)BATCH * TSEQ * EMB;      // 4 Mi elems
  const size_t nWh  = (size_t)NH * EMB * HD;           // 1 Mi elems
  const size_t nWp  = (size_t)EMB * EMB;               // 1 Mi elems
  const size_t nQKV = (size_t)BATCH * NH * TSEQ * HD;  // 4 Mi elems

  bf16* xq  = (bf16*)alloc(nAct * 2);
  bf16* xm  = (bf16*)alloc(nAct * 2);
  bf16* wqt = (bf16*)alloc(nWh * 2);          // [H, D, E]
  bf16* wkt = (bf16*)alloc(nWh * 2);
  bf16* wvt = (bf16*)alloc(nWh * 2);
  bf16* wpt = (bf16*)alloc(nWp * 2);          // [E, F]
  bf16* Qb  = (bf16*)alloc(nQKV * 2);         // [BH, T, D]
  bf16* Kb  = (bf16*)alloc(nQKV * 2);
  bf16* Vt  = (bf16*)alloc(nQKV * 2 + 4096);  // [BH, D, T] (+tail over-read pad)
  bf16* Ob  = (bf16*)alloc(nAct * 2);
  float* sc = (float*)alloc(((size_t)TSEQ * TSEQ + 64) * 4);  // (+tail pad)
  float* th = (float*)alloc((size_t)TSEQ * 4);
  (void)ws_size; (void)in_sizes; (void)n_in; (void)out_size;

  pack_bf16_kernel<<<(int)((nAct + 255) / 256), 256, 0, stream>>>(index,  xq, (int)nAct);
  pack_bf16_kernel<<<(int)((nAct + 255) / 256), 256, 0, stream>>>(memory, xm, (int)nAct);
  pack_transpose_kernel<<<(int)((nWh + 255) / 256), 256, 0, stream>>>(Wq, wqt, NH, EMB, HD);
  pack_transpose_kernel<<<(int)((nWh + 255) / 256), 256, 0, stream>>>(Wk, wkt, NH, EMB, HD);
  pack_transpose_kernel<<<(int)((nWh + 255) / 256), 256, 0, stream>>>(Wv, wvt, NH, EMB, HD);
  pack_transpose_kernel<<<(int)((nWp + 255) / 256), 256, 0, stream>>>(Wp, wpt, 1, NH * HD, EMB);

  qkv_wmma_kernel<<<dim3(TSEQ / 16, 1, BATCH * NH * 3), 32, 0, stream>>>(
      xq, xm, wqt, wkt, wvt, Qb, Kb, Vt);

  for (int bh = 0; bh < BATCH * NH; ++bh) {
    scores_wmma_kernel<<<dim3(TSEQ / 16, TSEQ / 64), 32, 0, stream>>>(Qb, Kb, sc, bh);
    topk_thresh_kernel<<<TSEQ, 128, 0, stream>>>(sc, th);
    softmax_av_kernel<<<TSEQ / 16, 256, 0, stream>>>(sc, th, Vt, Ob, bh);
  }

  proj_wmma_kernel<<<dim3(BATCH * TSEQ / 16, EMB / 64), 32, 0, stream>>>(Ob, wpt, bp, out);
}
